// LocalPixelRelationConv_2791728742719
// MI455X (gfx1250) — matrix operand order repaired and checked by reading, not compiled, not run on hardware
//
#include <hip/hip_runtime.h>
#include <hip/hip_bf16.h>

typedef __attribute__((ext_vector_type(16))) __bf16 v16bf;
typedef __attribute__((ext_vector_type(8)))  float  v8f;

#define HH 128
#define WW 128
#define CC 64
#define K2 9
#define TILE 16
#define PTS 18            // padded tile size (16 + 2)
#define XS_STRIDE 72      // bf16 elems per pixel row in LDS (64 + pad)
#define FS_STRIDE 68      // f32 elems per pixel row in LDS (64 + pad)

__global__ __launch_bounds__(256)
void lprc_kernel(const float* __restrict__ x,
                 const float* __restrict__ feat,
                 const float* __restrict__ conv_w,
                 const float* __restrict__ bn_gamma,
                 const float* __restrict__ bn_beta,
                 const float* __restrict__ bn_mean,
                 const float* __restrict__ bn_var,
                 float* __restrict__ out_rel,   // [B,9,H,W]
                 float* __restrict__ out_res)   // [B,C,H,W]
{
    __shared__ __bf16 xs[PTS * PTS * XS_STRIDE];   // 46656 B  x tile, bf16, pixel-major
    __shared__ float  fs[PTS * PTS * FS_STRIDE];   // 88128 B  feat tile, f32, pixel-major
    __shared__ __bf16 Amat[K2 * 16 * CC];          // 18432 B  per-tap 16x64 A matrices
    __shared__ float  wkLDS[256 * 12];             // 12288 B  softmax weights per pixel
    __shared__ float  sS[16];
    __shared__ float  sBb[16];

    const int tid  = threadIdx.x;
    const int lane = tid & 31;
    const int wave = tid >> 5;
    const int b    = blockIdx.z;
    const int gy0  = blockIdx.y * TILE;
    const int gx0  = blockIdx.x * TILE;

    // BN folded to scale/shift
    if (tid < K2) {
        float s  = bn_gamma[tid] * rsqrtf(bn_var[tid] + 1e-5f);
        sS[tid]  = s;
        sBb[tid] = bn_beta[tid] - bn_mean[tid] * s;
    }

    // conv_w [9][64][3][3] -> Amat[tap][m(16)][c(64)] bf16, rows m>=9 zero
    for (int i = tid; i < K2 * 16 * CC; i += 256) {
        int tap = i >> 10;        // /1024
        int rem = i & 1023;
        int m   = rem >> 6;
        int c   = rem & 63;
        float wv = (m < K2) ? conv_w[(m * CC + c) * K2 + tap] : 0.f;
        Amat[i] = (__bf16)wv;
    }

    // Replicate-pad clamped tile loads: lanes cover a row of 18 px (coalesced in gx)
    for (int row = (tid >> 5); row < CC * PTS; row += 8) {
        if (lane < PTS) {
            int c  = row / PTS;
            int py = row - c * PTS;
            int gy = min(max(gy0 + py  - 1, 0), HH - 1);
            int gx = min(max(gx0 + lane - 1, 0), WW - 1);
            size_t gidx = ((size_t)((b * CC + c) * HH + gy)) * WW + gx;
            float xv = x[gidx];
            xs[(py * PTS + lane) * XS_STRIDE + c] = (__bf16)xv;
            fs[(py * PTS + lane) * FS_STRIDE + c] = feat[gidx];
        }
    }
    __syncthreads();

    // ---- conv as implicit GEMM: 9 taps x 2 K-chunks of 32 channels ----
    v8f acc0 = {};
    v8f acc1 = {};
    const int  n     = lane & 15;
    const bool hi    = lane >= 16;
    const int  aKoff = hi ? 16 : 0;   // A in this code's contiguous k order: lanes>=16 hold K 16..31
    const int  bKoff = hi ? 16 : 0;   // B, same contiguous order as A: lanes>=16 hold K 16..31 of column n
    const int  py0   = 2 * wave;

    union FragU { v16bf v; uint4 q[2]; };

    #pragma unroll
    for (int tap = 0; tap < 9; ++tap) {
        const int dy = tap / 3, dx = tap % 3;
        #pragma unroll
        for (int kc = 0; kc < 2; ++kc) {
            FragU a;
            const __bf16* ap = &Amat[(tap * 16 + n) * CC + kc * 32 + aKoff];
            a.q[0] = *reinterpret_cast<const uint4*>(ap);        // K +0..7
            a.q[1] = *reinterpret_cast<const uint4*>(ap + 8);    // K +8..15

            FragU b0, b1;
            const __bf16* bp0 = &xs[((py0 + 0 + dy) * PTS + (n + dx)) * XS_STRIDE + kc * 32 + bKoff];
            b0.q[0] = *reinterpret_cast<const uint4*>(bp0);
            b0.q[1] = *reinterpret_cast<const uint4*>(bp0 + 8);
            const __bf16* bp1 = &xs[((py0 + 1 + dy) * PTS + (n + dx)) * XS_STRIDE + kc * 32 + bKoff];
            b1.q[0] = *reinterpret_cast<const uint4*>(bp1);
            b1.q[1] = *reinterpret_cast<const uint4*>(bp1 + 8);

            acc0 = __builtin_amdgcn_wmma_f32_16x16x32_bf16(false, a.v, false, b0.v,
                                                           (short)0, acc0, false, false);
            acc1 = __builtin_amdgcn_wmma_f32_16x16x32_bf16(false, a.v, false, b1.v,
                                                           (short)0, acc1, false, false);
        }
    }

    // ---- BN + pixel_relation store + per-pixel softmax over 9 taps ----
    #pragma unroll
    for (int r2 = 0; r2 < 2; ++r2) {
        const int py = py0 + r2;
        const int gy = gy0 + py;
        const int gx = gx0 + n;
        v8f acc = r2 ? acc1 : acc0;
        float bn[8];
        #pragma unroll
        for (int v = 0; v < 8; ++v) {
            int o = v + (hi ? 8 : 0);           // D layout: lanes>=16 hold rows M=8..15
            float bv = 0.f;
            if (o < K2) {
                bv = acc[v] * sS[o] + sBb[o];
                out_rel[((size_t)(b * K2 + o) * HH + gy) * WW + gx] = bv;
            }
            bn[v] = bv;
        }
        // channel 8 lives in partner lane (lane+16, VGPR0)
        float ch8 = __shfl_xor(bn[0], 16, 32);
        if (!hi) {
            float mx = ch8;
            #pragma unroll
            for (int v = 0; v < 8; ++v) mx = fmaxf(mx, bn[v]);
            float e8 = __expf(ch8 - mx);
            float e[8];
            float s = e8;
            #pragma unroll
            for (int v = 0; v < 8; ++v) { e[v] = __expf(bn[v] - mx); s += e[v]; }
            float inv = 1.f / s;
            float* wp = &wkLDS[(py * 16 + n) * 12];
            #pragma unroll
            for (int v = 0; v < 8; ++v) wp[v] = e[v] * inv;
            wp[8] = e8 * inv;
        }
    }
    __syncthreads();

    // ---- aggregation: res[c,pix] = sum_k wk[k,pix] * feat[c, pix shifted by tap k] ----
    {
        const int px2 = tid & 15, py2 = tid >> 4;
        const int gy = gy0 + py2, gx = gx0 + px2;
        float wv[9];
        #pragma unroll
        for (int k = 0; k < 9; ++k) wv[k] = wkLDS[tid * 12 + k];

        #pragma unroll
        for (int cb = 0; cb < CC; cb += 32) {
            float r[32];
            #pragma unroll
            for (int c = 0; c < 32; ++c) r[c] = 0.f;
            #pragma unroll
            for (int k = 0; k < 9; ++k) {
                const float* fb = &fs[((py2 + k / 3) * PTS + px2 + (k % 3)) * FS_STRIDE + cb];
                float w = wv[k];
                #pragma unroll
                for (int c = 0; c < 32; ++c) r[c] = fmaf(w, fb[c], r[c]);
            }
            #pragma unroll
            for (int c = 0; c < 32; ++c)
                out_res[((size_t)(b * CC + cb + c) * HH + gy) * WW + gx] = r[c];
        }
    }
}

extern "C" void kernel_launch(void* const* d_in, const int* in_sizes, int n_in,
                              void* d_out, int out_size, void* d_ws, size_t ws_size,
                              hipStream_t stream) {
    const float* x      = (const float*)d_in[0];
    const float* feat   = (const float*)d_in[1];
    const float* conv_w = (const float*)d_in[2];
    const float* gmm    = (const float*)d_in[3];
    const float* beta   = (const float*)d_in[4];
    const float* mean   = (const float*)d_in[5];
    const float* var    = (const float*)d_in[6];

    float* out     = (float*)d_out;
    float* out_rel = out;                                  // [8,9,128,128]
    float* out_res = out + (size_t)8 * K2 * HH * WW;       // [8,64,128,128]

    dim3 grid(WW / TILE, HH / TILE, 8);
    lprc_kernel<<<grid, 256, 0, stream>>>(x, feat, conv_w, gmm, beta, mean, var,
                                          out_rel, out_res);
}
